// LambdaLayer_64888365908313
// MI455X (gfx1250) — matrix operand order repaired and checked by reading, not compiled, not run on hardware
//
#include <hip/hip_runtime.h>
#include <hip/hip_bf16.h>
#include <stdint.h>

// ---------------- problem constants ----------------
#define BB    32
#define NN    4096
#define CC    512
#define HEADS 4
#define KK    16
#define UU    4
#define VV    128
#define COLS  640          // 64 q | 64 k | 512 v
#define KS    7
#define EPS   1e-3f

typedef __attribute__((ext_vector_type(16))) _Float16 v16h;
typedef __attribute__((ext_vector_type(8)))  _Float16 v8h;
typedef __attribute__((ext_vector_type(8)))  float    v8f;
typedef __attribute__((ext_vector_type(4)))  unsigned int u32x4;
typedef __attribute__((ext_vector_type(8)))  int          i32x8;
typedef __attribute__((ext_vector_type(4)))  int          i32x4;

union F16x16 { v16h v; v8h h[2]; _Float16 e[16]; };

static __device__ inline v8f vzero8() {
    v8f z;
    for (int i = 0; i < 8; ++i) z[i] = 0.0f;
    return z;
}

// =====================================================================
// K0: weight prep.
//  wT[col][c]   : f16, col-major-transposed concat of [wq|wk|wv] (640 x 512)
//  wposT[k][su] : f16, 16 x 32 (su = s*4+u, padded 28->32 with zeros)
// =====================================================================
__global__ __launch_bounds__(256) void k_prep(
    const float* __restrict__ wq, const float* __restrict__ wk,
    const float* __restrict__ wv, const float* __restrict__ pos_w,
    _Float16* __restrict__ wT, _Float16* __restrict__ wposT)
{
    const int j = blockIdx.x;  // 0..639
    for (int c = threadIdx.x; c < CC; c += 256) {
        float v;
        if (j < 64)        v = wq[c * 64 + j];
        else if (j < 128)  v = wk[c * 64 + (j - 64)];
        else               v = wv[c * 512 + (j - 128)];
        wT[(long)j * CC + c] = (_Float16)v;
    }
    if (blockIdx.x == 0) {
        for (int i = threadIdx.x; i < 16 * 32; i += 256) {
            int k = i >> 5, su = i & 31;
            float v = 0.0f;
            if (su < KS * UU) v = pos_w[su * KK + k];   // pos_w[(s*U+u)*K + k]
            wposT[i] = (_Float16)v;
        }
    }
}

// =====================================================================
// K1: fused projection GEMM (131072 x 640 x 512) with WMMA f16 and a
// software-pipelined K loop (next chunk's A/B fragments load while the
// current chunk's WMMAs run). Fused epilogue: LN(q), LN(v), k^T store.
// One block = 16 rows x 640 cols; 8 waves x 5 column tiles each.
// =====================================================================
__global__ __launch_bounds__(256) void k_proj(
    const float* __restrict__ x, const _Float16* __restrict__ wT,
    const float* __restrict__ gq, const float* __restrict__ bq,
    const float* __restrict__ gv, const float* __restrict__ bv,
    float* __restrict__ qbuf,        // (B*N, 64) f32
    float* __restrict__ kraw,        // (B, 64, N) f32   (transposed)
    _Float16* __restrict__ vn)       // (B, U, N, V) f16
{
    __shared__ __align__(32) _Float16 xa[16][CC];   // 16 KB
    __shared__ float raw[16][COLS];                 // 40 KB
    __shared__ float red[256];
    __shared__ float red2[256];

    const int tid  = threadIdx.x;
    const int wave = tid >> 5, lane = tid & 31;
    const int grp  = lane >> 4, l16 = lane & 15;
    const long rowBase = (long)blockIdx.x * 16;

    // ---- stage x tile (fp32 -> f16) into LDS ----
    for (int i = tid; i < 16 * CC; i += 256) {
        int r = i >> 9, c = i & (CC - 1);
        xa[r][c] = (_Float16)x[(rowBase + r) * CC + c];
    }
    __syncthreads();

    // ---- WMMA main loop, software pipelined ----
    v8f acc[5];
    for (int t = 0; t < 5; ++t) acc[t] = vzero8();

    const int colBase = wave * 5 * 16 + l16;

    F16x16 a_cur, b_cur[5], a_nxt, b_nxt[5];
    a_cur.h[0] = *(const v8h*)&xa[l16][16 * grp];
    a_cur.h[1] = *(const v8h*)&xa[l16][16 * grp + 8];
    for (int t = 0; t < 5; ++t)
        b_cur[t].v = *(const v16h*)&wT[(long)(colBase + t * 16) * CC + 16 * grp];

    for (int k0 = 0; k0 < CC; k0 += 32) {
        const int k1 = k0 + 32;
        if (k1 < CC) {
            a_nxt.h[0] = *(const v8h*)&xa[l16][k1 + 16 * grp];
            a_nxt.h[1] = *(const v8h*)&xa[l16][k1 + 16 * grp + 8];
            for (int t = 0; t < 5; ++t)
                b_nxt[t].v = *(const v16h*)&wT[(long)(colBase + t * 16) * CC + k1 + 16 * grp];
        }
        for (int t = 0; t < 5; ++t)
            acc[t] = __builtin_amdgcn_wmma_f32_16x16x32_f16(
                false, a_cur.v, false, b_cur[t].v, (short)0, acc[t], false, false);
        if (k1 < CC) {
            a_cur = a_nxt;
            for (int t = 0; t < 5; ++t) b_cur[t] = b_nxt[t];
        }
    }

    // ---- dump accumulators to LDS (D: M = r + 8*grp, N = l16) ----
    for (int t = 0; t < 5; ++t) {
        const int col = (wave * 5 + t) * 16 + l16;
        for (int r = 0; r < 8; ++r) raw[r + 8 * grp][col] = acc[t][r];
    }
    __syncthreads();

    // ---- epilogue: 16 threads per row ----
    const int r = tid >> 4, i = tid & 15;
    const long rg = rowBase + r;
    const int  bb = (int)(rg >> 12);        // N = 4096
    const int  n  = (int)(rg & (NN - 1));

    // q LayerNorm over cols [0,64)
    float s = 0.f, s2 = 0.f;
    for (int j = 0; j < 4; ++j) { float v = raw[r][i * 4 + j]; s += v; s2 += v * v; }
    red[tid] = s; red2[tid] = s2; __syncthreads();
    if (i < 8) { red[tid] += red[tid + 8]; red2[tid] += red2[tid + 8]; } __syncthreads();
    if (i < 4) { red[tid] += red[tid + 4]; red2[tid] += red2[tid + 4]; } __syncthreads();
    if (i < 2) { red[tid] += red[tid + 2]; red2[tid] += red2[tid + 2]; } __syncthreads();
    if (i < 1) { red[tid] += red[tid + 1]; red2[tid] += red2[tid + 1]; } __syncthreads();
    const float qmu  = red[r << 4] * (1.f / 64.f);
    const float qvar = red2[r << 4] * (1.f / 64.f) - qmu * qmu;
    const float qrs  = rsqrtf(qvar + EPS);
    __syncthreads();

    for (int j = 0; j < 4; ++j) {
        int c = i * 4 + j;
        qbuf[rg * 64 + c] = (raw[r][c] - qmu) * qrs * gq[c] + bq[c];
    }
    // k: plain copy, transposed to (B, 64, N)
    for (int j = 0; j < 4; ++j) {
        int c = i * 4 + j;
        kraw[((long)bb * 64 + c) * NN + n] = raw[r][64 + c];
    }

    // v LayerNorm over cols [128,640) (512 ch), lane owns contiguous 32
    s = 0.f; s2 = 0.f;
    for (int j = 0; j < 32; ++j) { float v = raw[r][128 + i * 32 + j]; s += v; s2 += v * v; }
    red[tid] = s; red2[tid] = s2; __syncthreads();
    if (i < 8) { red[tid] += red[tid + 8]; red2[tid] += red2[tid + 8]; } __syncthreads();
    if (i < 4) { red[tid] += red[tid + 4]; red2[tid] += red2[tid + 4]; } __syncthreads();
    if (i < 2) { red[tid] += red[tid + 2]; red2[tid] += red2[tid + 2]; } __syncthreads();
    if (i < 1) { red[tid] += red[tid + 1]; red2[tid] += red2[tid + 1]; } __syncthreads();
    const float vmu  = red[r << 4] * (1.f / 512.f);
    const float vvar = red2[r << 4] * (1.f / 512.f) - vmu * vmu;
    const float vrs  = rsqrtf(vvar + EPS);
    __syncthreads();

    {
        const int cc0 = i * 32;            // within one u-block (32 | 128)
        const int u   = cc0 >> 7;
        const int vvb = cc0 & (VV - 1);
        const long base = (((long)bb * UU + u) * NN + n) * VV + vvb;
        for (int j = 0; j < 32; ++j) {
            int cc = cc0 + j;
            float v = (raw[r][128 + cc] - vmu) * vrs * gv[cc] + bv[cc];
            vn[base + j] = (_Float16)v;
        }
    }
}

// =====================================================================
// K2: softmax over n for each (b, u*K+k) row of kraw (B,64,N) -> ksm f16
// =====================================================================
__global__ __launch_bounds__(256) void k_softmax(
    const float* __restrict__ kraw, _Float16* __restrict__ ksm)
{
    __shared__ float red[256];
    const long base = (long)blockIdx.x * NN;
    const int  t = threadIdx.x;
    float v[16];
    float m = -1e30f;
    for (int i = 0; i < 16; ++i) { v[i] = kraw[base + t + 256 * i]; m = fmaxf(m, v[i]); }
    red[t] = m; __syncthreads();
    for (int st = 128; st >= 1; st >>= 1) {
        if (t < st) red[t] = fmaxf(red[t], red[t + st]);
        __syncthreads();
    }
    m = red[0]; __syncthreads();
    float s = 0.f;
    for (int i = 0; i < 16; ++i) { v[i] = __expf(v[i] - m); s += v[i]; }
    red[t] = s; __syncthreads();
    for (int st = 128; st >= 1; st >>= 1) {
        if (t < st) red[t] += red[t + st];
        __syncthreads();
    }
    const float inv = 1.0f / red[0];
    for (int i = 0; i < 16; ++i) ksm[base + t + 256 * i] = (_Float16)(v[i] * inv);
}

// =====================================================================
// K3: content lambda  lc[b,k,v] = sum_{u,n} ksm[b,u,k,n] * vn[b,u,n,v]
// One block per b; wave w owns the 16-wide vv column tile w.
// vn chunks (64 x 128 f16 = 16 KB) are DMA'd into a double-buffered LDS
// tile by the Tensor Data Mover; next chunk's DMA overlaps current WMMAs.
// =====================================================================
__global__ __launch_bounds__(256) void k_lc(
    const _Float16* __restrict__ ksm, const _Float16* __restrict__ vn,
    float* __restrict__ lc)
{
    __shared__ __align__(16) _Float16 vtile[2][64][VV];   // 2 x 16 KB

    const int b    = blockIdx.x;
    const int tid  = threadIdx.x;
    const int wave = tid >> 5, lane = tid & 31;
    const int grp  = lane >> 4, l16 = lane & 15;
    const int col  = wave * 16 + l16;                 // vv

    const _Float16* vb = vn + ((long)b * UU) * NN * VV;   // this batch's slab

    // TDM 1-D copy: 16 KB (2048 x 8B units) from vn chunk -> vtile[buf]
    auto issue_tdm = [&](int c, int buf) {
        const unsigned long long ga = (unsigned long long)(size_t)
            (vb + (((long)(c >> 6)) * NN + (long)(c & 63) * 64) * VV);
        const unsigned lds = (unsigned)(size_t)(&vtile[buf][0][0]);
        u32x4 g0;
        g0[0] = 1u;                                            // count=1 (valid)
        g0[1] = lds;                                           // LDS byte addr
        g0[2] = (unsigned)ga;                                  // global addr lo
        g0[3] = (unsigned)((ga >> 32) & 0x01FFFFFFu) | 0x80000000u; // hi | type=2
        i32x8 g1;
        g1[0] = (int)(3u << 16);            // data_size = 8 bytes
        g1[1] = (int)(2048u << 16);         // tensor_dim0[15:0] (=2048 units)
        g1[2] = (int)(1u << 16);            // tensor_dim0 hi=0 | tensor_dim1=1
        g1[3] = (int)(2048u << 16);         // tensor_dim1 hi=0 | tile_dim0=2048
        g1[4] = 1;                          // tile_dim1=1, tile_dim2=0
        g1[5] = 2048;                       // tensor_dim0_stride lo
        g1[6] = 0;                          // stride0 hi | stride1 lo
        g1[7] = 0;
        i32x4 gz; gz[0] = 0; gz[1] = 0; gz[2] = 0; gz[3] = 0;
        i32x8 gz8;
        for (int z = 0; z < 8; ++z) gz8[z] = 0;
        __builtin_amdgcn_tensor_load_to_lds(g0, g1, gz, gz, gz8, 0);
    };

    v8f acc = vzero8();

    if (wave == 0) {
        issue_tdm(0, 0);
        __builtin_amdgcn_s_wait_tensorcnt(0);
    }
    __syncthreads();

    for (int c = 0; c < 256; ++c) {                 // 4 u * 64 n-chunks
        const int buf = c & 1;
        if ((wave == 0) && (c + 1 < 256)) issue_tdm(c + 1, buf ^ 1);

        const int u  = c >> 6;
        const int n0 = (c & 63) * 64;
        const _Float16* arow = ksm + ((long)b * 64 + u * 16 + l16) * NN;

        for (int kk = 0; kk < 64; kk += 32) {
            F16x16 a;
            a.h[0] = *(const v8h*)&arow[n0 + kk + 16 * grp];
            a.h[1] = *(const v8h*)&arow[n0 + kk + 16 * grp + 8];
            F16x16 bf;   // B[n, vv]: K = kk + 16*grp + j at fixed col
            for (int j = 0; j < 16; ++j)
                bf.e[j] = vtile[buf][kk + 16 * grp + j][col];
            acc = __builtin_amdgcn_wmma_f32_16x16x32_f16(
                false, a.v, false, bf.v, (short)0, acc, false, false);
        }
        if (wave == 0) __builtin_amdgcn_s_wait_tensorcnt(0);
        __syncthreads();
    }
    for (int r = 0; r < 8; ++r)
        lc[((long)b * KK + r + 8 * grp) * VV + col] = acc[r];
}

// =====================================================================
// K4: fused positional-lambda conv (im2col WMMA, lp kept in LDS) +
// final per-position contraction  out = q . (lc + lp) + (q . pos_b)
// Block tile: (b, 32 positions, 16 vv). q tile staged by async-to-LDS.
// =====================================================================
__global__ __launch_bounds__(256) void k_out(
    const float* __restrict__ qbuf, const _Float16* __restrict__ vn,
    const float* __restrict__ lc, const _Float16* __restrict__ wposT,
    const float* __restrict__ pos_b, float* __restrict__ out)
{
    __shared__ __align__(32) _Float16 vwin[38][UU][16];   // 4.75 KB
    __shared__ __align__(16) float qtile[32][64];         // 8 KB
    __shared__ float lctile[16][16];                      // 1 KB
    __shared__ float lpt[32][16][16];                     // 32 KB  [n][k][vv]

    const int b   = blockIdx.z;
    const int n0  = blockIdx.y * 32;
    const int vv0 = blockIdx.x * 16;
    const int tid  = threadIdx.x;
    const int wave = tid >> 5, lane = tid & 31;
    const int grp  = lane >> 4, l16 = lane & 15;

    // ---- async stage q tile: 8 KB fully contiguous (2 x b128 per thread) ----
    {
        const unsigned long long src =
            (unsigned long long)(size_t)(qbuf + ((long)b * NN + n0) * 64);
        const unsigned lbase = (unsigned)(size_t)(&qtile[0][0]);
        const unsigned off0  = (unsigned)tid * 16u;
        asm volatile("global_load_async_to_lds_b128 %0, %1, %2"
                     :: "v"(lbase + off0), "v"(off0), "s"(src) : "memory");
        asm volatile("global_load_async_to_lds_b128 %0, %1, %2"
                     :: "v"(lbase + off0 + 4096u), "v"(off0 + 4096u), "s"(src)
                     : "memory");
    }

    // ---- stage v window (SAME zero padding) + lc tile ----
    for (int i = tid; i < 38 * 64; i += 256) {
        int nl = i >> 6, uv = i & 63, u = uv >> 4, vvl = uv & 15;
        int n = n0 - 3 + nl;
        _Float16 val = (_Float16)0.0f;
        if (n >= 0 && n < NN)
            val = vn[(((long)b * UU + u) * NN + n) * VV + vv0 + vvl];
        vwin[nl][u][vvl] = val;
    }
    {
        int k = tid >> 4, vv = tid & 15;
        lctile[k][vv] = lc[((long)b * KK + k) * VV + vv0 + vv];
    }
    asm volatile("s_wait_asynccnt 0" ::: "memory");
    __syncthreads();

    // ---- conv as GEMM: M = (vv,n) rows, N = k (16), K = (s,u) padded 32 ----
    F16x16 bw;   // B 32x16: col = k = l16, K = su = 16*grp + j (contiguous)
    bw.v = *(const v16h*)&wposT[l16 * 32 + 16 * grp];

    for (int tt = 0; tt < 4; ++tt) {
        const int mt   = wave * 4 + tt;     // 32 M-tiles total
        const int vvl  = mt >> 1;
        const int noff = (mt & 1) * 16;
        const int nl   = noff + l16;        // local n for this A row
        F16x16 a;
        for (int j = 0; j < 16; ++j) {
            int su = 16 * grp + j;
            _Float16 val = (_Float16)0.0f;
            if (su < KS * UU) {
                int s = su >> 2, u = su & 3;
                val = vwin[nl + s][u][vvl];
            }
            a.e[j] = val;
        }
        v8f acc = vzero8();
        acc = __builtin_amdgcn_wmma_f32_16x16x32_f16(
            false, a.v, false, bw.v, (short)0, acc, false, false);
        for (int r = 0; r < 8; ++r)
            lpt[noff + r + 8 * grp][l16][vvl] = acc[r];   // [n][k][vv]
    }
    __syncthreads();

    // ---- final: out[n,h,vv] = sum_k q[n,h,k]*(lc[k,vv]+lp[n,k,vv]) + qb ----
    const int pair = tid >> 1;      // 128 (n,h) pairs
    const int half = tid & 1;       // vv half
    const int n = pair >> 2, h = pair & 3;
    float o[8];
    for (int j = 0; j < 8; ++j) o[j] = 0.0f;
    float qb = 0.0f;
    for (int k = 0; k < KK; ++k) {
        const float qv = qtile[n][h * 16 + k];
        qb += qv * pos_b[k];
        for (int j = 0; j < 8; ++j) {
            int vv = half * 8 + j;
            o[j] += qv * (lctile[k][vv] + lpt[n][k][vv]);
        }
    }
    const long ob = ((long)b * NN + n0 + n) * (HEADS * VV) + h * VV + vv0 + half * 8;
    for (int j = 0; j < 8; ++j) out[ob + j] = o[j] + qb;
}

// =====================================================================
// launch
// =====================================================================
extern "C" void kernel_launch(void* const* d_in, const int* in_sizes, int n_in,
                              void* d_out, int out_size, void* d_ws, size_t ws_size,
                              hipStream_t stream)
{
    const float* x     = (const float*)d_in[0];
    const float* wq    = (const float*)d_in[1];
    const float* wk    = (const float*)d_in[2];
    const float* wv    = (const float*)d_in[3];
    const float* gq    = (const float*)d_in[4];
    const float* bq    = (const float*)d_in[5];
    const float* gv    = (const float*)d_in[6];
    const float* bv    = (const float*)d_in[7];
    const float* pos_w = (const float*)d_in[8];
    const float* pos_b = (const float*)d_in[9];
    float* outp = (float*)d_out;

    uint8_t* ws = (uint8_t*)d_ws;
    size_t off = 0;
    auto take = [&](size_t bytes) -> void* {
        void* p = ws + off;
        off = (off + bytes + 255) & ~(size_t)255;
        return p;
    };
    _Float16* wT    = (_Float16*)take((size_t)COLS * CC * sizeof(_Float16));   // 640 KB
    _Float16* wposT = (_Float16*)take((size_t)16 * 32 * sizeof(_Float16));     // 1 KB
    float*    qbuf  = (float*)take((size_t)BB * NN * 64 * sizeof(float));      // 32 MB
    float*    kraw  = (float*)take((size_t)BB * 64 * NN * sizeof(float));      // 32 MB
    _Float16* ksm   = (_Float16*)take((size_t)BB * 64 * NN * sizeof(_Float16));// 16 MB
    _Float16* vn    = (_Float16*)take((size_t)BB * UU * NN * VV * sizeof(_Float16)); // 128 MB
    float*    lcb   = (float*)take((size_t)BB * KK * VV * sizeof(float));      // 256 KB
    (void)ws_size; (void)in_sizes; (void)n_in; (void)out_size;

    k_prep<<<COLS, 256, 0, stream>>>(wq, wk, wv, pos_w, wT, wposT);
    k_proj<<<(BB * NN) / 16, 256, 0, stream>>>(x, wT, gq, bq, gv, bv, qbuf, kraw, vn);
    k_softmax<<<BB * 64, 256, 0, stream>>>(kraw, ksm);
    k_lc<<<BB, 256, 0, stream>>>(ksm, vn, lcb);
    k_out<<<dim3(VV / 16, NN / 32, BB), 256, 0, stream>>>(qbuf, vn, lcb, wposT, pos_b, outp);
}
